// MACE_45981919871462
// MI455X (gfx1250) — matrix operand order repaired and checked by reading, not compiled, not run on hardware
//
// MACE-style GNN for MI455X (gfx1250, wave32).
// Edge radial MLP runs on v_wmma_f32_16x16x32_f16 (16-edge tiles, weights in LDS);
// message scatter fused behind it so the E x 96 MLP output never hits HBM.
// The N x 9 x 32 accumulator (57.6 MB) is L2-resident (192 MB L2), so the
// segment-sum atomics resolve in-cache.
// v3: fragment assembly via wide LDS ops -- A frags are 2x ds_load_b128
// (contiguous K runs per the ISA layout), B frags are one v16h load from
// weights pre-swizzled into per-lane fragment order at staging time.
// No per-element guards -> no exec-mask dances.
#include <hip/hip_runtime.h>
#include <math.h>

#define NN    50000
#define EE    800000
#define ZZ    10
#define CC    32
#define NBF   8
#define HH    64
#define MLPHH 16
#define NSH   9
#define RMAX  5.0f

typedef __attribute__((ext_vector_type(16))) _Float16 v16h;
typedef __attribute__((ext_vector_type(8)))  _Float16 v8h;
typedef __attribute__((ext_vector_type(8)))  float    v8f;

__device__ __forceinline__ int lmap_of(int lm) { return (lm == 0) ? 0 : ((lm < 4) ? 1 : 2); }

// fast SiLU: x * sigmoid(x) with v_rcp_f32 (avoids v_div_scale/Newton chains)
__device__ __forceinline__ float fast_silu(float x) {
  return x * __builtin_amdgcn_rcpf(1.0f + __expf(-x));
}

__device__ __forceinline__ float warp_sum(float v) {
#pragma unroll
  for (int m = 16; m >= 1; m >>= 1) v += __shfl_xor(v, m, 32);
  return v;
}

// ---- WMMA fragment builders per CDNA5 ISA section 7.12.2 (wave32) ----
// 16-bit A 16x32: lane (m = lane%16, hv = lane/16) holds K runs
// [kbase + hv*8, +8) and [kbase + 16 + hv*8, +8): two contiguous 16B chunks.
template<int LDH>
__device__ __forceinline__ v16h load_a_frag(const _Float16* src, int kbase) {
  const int lane = threadIdx.x & 31;
  const int m = lane & 15;
  const int hv = lane >> 4;
  const _Float16* p = src + m * LDH + kbase + hv * 8;
  const v8h lo = *(const v8h*)(p);
  const v8h hi = *(const v8h*)(p + 16);
  v16h a;
#pragma unroll
  for (int j = 0; j < 8; ++j) { a[j] = lo[j]; a[j + 8] = hi[j]; }
  return a;
}

// B fragments come from weights pre-swizzled into per-lane fragment order:
// one contiguous 32B v16h load per (ktile, ntile).
__device__ __forceinline__ v16h load_b_frag(const _Float16* sWst, int tileIdx) {
  const int lane = threadIdx.x & 31;
  return *(const v16h*)(sWst + (tileIdx * 32 + lane) * 16);
}

// Stage W[KSRC][NOUT] (f32, row-major) into LDS in fragment order, zero-padding
// K up to KDIM. Element (tile=ntIdx*KT+kt, lane, j) <- W[kt*32 + hv*8 + (j&7) + (j>>3)*16][ntIdx*16 + n].
template<int KDIM, int NOUT, int KSRC>
__device__ void stage_weight(const float* __restrict__ W, _Float16* sWst) {
  constexpr int KT = KDIM / 32;
  constexpr int total = KDIM * NOUT;
  for (int idx = threadIdx.x; idx < total; idx += blockDim.x) {
    const int j = idx & 15;
    const int lane = (idx >> 4) & 31;
    const int tile = idx >> 9;
    const int ntIdx = tile / KT, kt = tile % KT;
    const int hv = lane >> 4, n = lane & 15;
    const int k = kt * 32 + hv * 8 + (j & 7) + ((j >> 3) << 4);
    const int col = ntIdx * 16 + n;
    sWst[idx] = (k < KSRC) ? (_Float16)W[k * NOUT + col] : (_Float16)0.0f;
  }
}

// One dense layer on a 16-row tile: out[16][NOUT] = act( in[16][KDIM] @ W[KDIM][NOUT] )
template<int KDIM, int NOUT, int LDIN, bool SILU>
__device__ __forceinline__ void mlp_layer(const _Float16* sIn, const _Float16* sWst,
                                          _Float16* sOut) {
  const int lane = threadIdx.x & 31;
  const int hv = lane >> 4;
  const int nn = lane & 15;
  constexpr int KT = KDIM / 32;
  v16h afr[KT];
#pragma unroll
  for (int kt = 0; kt < KT; ++kt) afr[kt] = load_a_frag<LDIN>(sIn, kt * 32);
#pragma unroll
  for (int ntIdx = 0; ntIdx < NOUT / 16; ++ntIdx) {
    v8f acc = {};
#pragma unroll
    for (int kt = 0; kt < KT; ++kt) {
      const v16h b = load_b_frag(sWst, ntIdx * KT + kt);
      acc = __builtin_amdgcn_wmma_f32_16x16x32_f16(false, afr[kt], false, b,
                                                   (short)0, acc, false, false);
    }
    // D layout: VGPR v -> row (v + hv*8), col = lane%16
#pragma unroll
    for (int v = 0; v < 8; ++v) {
      float x = acc[v];
      if (SILU) x = fast_silu(x);
      sOut[(v + hv * 8) * 96 + ntIdx * 16 + nn] = (_Float16)x;
    }
  }
}

// ---------------- node init: energy = attrs@atomic_E ; h = attrs@W_embed ----------------
__global__ void k_node_init(const float* __restrict__ na, const float* __restrict__ Wemb,
                            const float* __restrict__ aE,
                            float* __restrict__ energy, float* __restrict__ hE) {
  const int n = blockIdx.x * blockDim.x + threadIdx.x;
  if (n >= NN) return;
  float e = 0.0f;
  for (int z = 0; z < ZZ; ++z) e += na[n * ZZ + z] * aE[z];
  energy[n] = e;
  for (int c = 0; c < CC; ++c) {
    float a = 0.0f;
    for (int z = 0; z < ZZ; ++z) a += na[n * ZZ + z] * Wemb[z * CC + c];
    hE[n * CC + c] = a;
  }
}

// ---------------- per-edge geometry: sh, radial basis, ZBL pair energy ----------------
__global__ void k_edge_geom(const float* __restrict__ na, const float* __restrict__ pos,
                            const int* __restrict__ snd, const int* __restrict__ rcv,
                            float* __restrict__ shb, float* __restrict__ Rb,
                            float* __restrict__ energy) {
  const int e = blockIdx.x * blockDim.x + threadIdx.x;
  if (e >= EE) return;
  const int s = snd[e], rc = rcv[e];
  const float dx = pos[rc * 3 + 0] - pos[s * 3 + 0];
  const float dy = pos[rc * 3 + 1] - pos[s * 3 + 1];
  const float dz = pos[rc * 3 + 2] - pos[s * 3 + 2];
  const float r = sqrtf(dx * dx + dy * dy + dz * dz + 1e-12f);
  const float inv_r = __builtin_amdgcn_rcpf(r);
  const float x = dx * inv_r, y = dy * inv_r, z = dz * inv_r;
  const float s3 = 1.7320508075688772f, s5 = 2.23606797749979f, s15 = 3.872983346207417f;
  float sh[NSH];
  sh[0] = 1.0f;            sh[1] = s3 * x;          sh[2] = s3 * y;
  sh[3] = s3 * z;          sh[4] = s15 * x * y;     sh[5] = s15 * y * z;
  sh[6] = 0.5f * s5 * (3.0f * z * z - 1.0f);
  sh[7] = s15 * x * z;     sh[8] = 0.5f * s15 * (x * x - y * y);
#pragma unroll
  for (int lm = 0; lm < NSH; ++lm) shb[e * NSH + lm] = sh[lm];

  const float u = r / RMAX;
  const float u2 = u * u;
  const float u5 = u2 * u2 * u;
  const float fcut = (u < 1.0f) ? (1.0f - 21.0f * u5 + 35.0f * u5 * u - 15.0f * u5 * u2) : 0.0f;
  const float pref = sqrtf(2.0f / RMAX);
#pragma unroll
  for (int nb = 0; nb < NBF; ++nb) {
    const float bessel = pref * __sinf((float)(nb + 1) * 3.14159265358979f * u) * inv_r;
    Rb[e * NBF + nb] = bessel * fcut;
  }
  float zi = 0.0f, zj = 0.0f;
  for (int zq = 0; zq < ZZ; ++zq) {
    zi += na[s * ZZ + zq] * (float)(zq + 1);
    zj += na[rc * ZZ + zq] * (float)(zq + 1);
  }
  const float a = 0.4685f / (powf(zi, 0.23f) + powf(zj, 0.23f));
  const float xx = r / a;
  const float phi = 0.1818f * __expf(-3.2f * xx) + 0.5099f * __expf(-0.9423f * xx)
                  + 0.2802f * __expf(-0.4029f * xx) + 0.02817f * __expf(-0.2016f * xx);
  const float vz = 14.3996f * zi * zj * inv_r * phi * fcut;
  atomicAdd(&energy[rc], 0.5f * vz);
}

// ---------------- h1 = h @ Wlin[i] (tiny GEMM, one thread per (n,c)) ----------------
__global__ void k_h_lin(const float* __restrict__ h, const float* __restrict__ Wlin,
                        float* __restrict__ h1) {
  const int t = blockIdx.x * blockDim.x + threadIdx.x;
  if (t >= NN * CC) return;
  const int n = t >> 5, c = t & 31;
  float acc = 0.0f;
#pragma unroll
  for (int cc = 0; cc < CC; ++cc) acc += h[n * CC + cc] * Wlin[cc * CC + c];
  h1[t] = acc;
}

// ---------------- edge MLP (WMMA) fused with message scatter ----------------
__global__ void __launch_bounds__(128)
k_edge_mlp(const float* __restrict__ Rb, const float* __restrict__ shb,
           const int* __restrict__ snd, const int* __restrict__ rcv,
           const float* __restrict__ h1,
           const float* __restrict__ Wr0, const float* __restrict__ Wr1,
           const float* __restrict__ Wr2, const float* __restrict__ Wr3,
           float* __restrict__ Abuf) {
  __shared__ _Float16 sW0st[32 * HH];       //  4 KB (K padded 8 -> 32)
  __shared__ _Float16 sW1st[HH * HH];       //  8 KB
  __shared__ _Float16 sW2st[HH * HH];       //  8 KB
  __shared__ _Float16 sW3st[HH * 3 * CC];   // 12 KB
  __shared__ _Float16 sBufX[4][16 * 32];    //  4 KB (layer-0 input, cols 8..31 == 0)
  __shared__ _Float16 sBufA[4][16 * 96];    // 12 KB (per-wave ping)
  __shared__ _Float16 sBufB[4][16 * 96];    // 12 KB (per-wave pong)

  stage_weight<32, HH, NBF>(Wr0, sW0st);
  stage_weight<HH, HH, HH>(Wr1, sW1st);
  stage_weight<HH, HH, HH>(Wr2, sW2st);
  stage_weight<HH, 3 * CC, HH>(Wr3, sW3st);

  const int wave = threadIdx.x >> 5;
  const int lane = threadIdx.x & 31;
  _Float16* bufX = sBufX[wave];
  _Float16* bufA = sBufA[wave];
  _Float16* bufB = sBufB[wave];

  // zero layer-0 input once; only cols 0..7 are rewritten per tile
  {
    unsigned* zx = (unsigned*)bufX;
#pragma unroll
    for (int t = lane; t < 16 * 32 / 2; t += 32) zx[t] = 0u;
  }
  __syncthreads();

  const int numTiles = (EE + 15) / 16;
  const int numGroups = (numTiles + 3) / 4;

  for (int tg = blockIdx.x; tg < numGroups; tg += gridDim.x) {
    const int tile = tg * 4 + wave;
    const int e0 = tile * 16;
    // Stage 16x8 radial features: lanes 0..15 each move one row (2x b128 global
    // loads -> 1x b128 LDS store).
    if (lane < 16) {
      const int e = e0 + lane;
      if (tile < numTiles && e < EE) {
        const float4 f0 = *(const float4*)(Rb + (size_t)e * NBF);
        const float4 f1 = *(const float4*)(Rb + (size_t)e * NBF + 4);
        v8h hx;
        hx[0] = (_Float16)f0.x; hx[1] = (_Float16)f0.y;
        hx[2] = (_Float16)f0.z; hx[3] = (_Float16)f0.w;
        hx[4] = (_Float16)f1.x; hx[5] = (_Float16)f1.y;
        hx[6] = (_Float16)f1.z; hx[7] = (_Float16)f1.w;
        *(v8h*)(bufX + lane * 32) = hx;
      }
    }
    __syncthreads();
    mlp_layer<32, HH, 32, true>(bufX, sW0st, bufA);      //  8 -> 64, SiLU (4 WMMA)
    __syncthreads();
    mlp_layer<HH, HH, 96, true>(bufA, sW1st, bufB);      // 64 -> 64, SiLU (8 WMMA)
    __syncthreads();
    mlp_layer<HH, HH, 96, true>(bufB, sW2st, bufA);      // 64 -> 64, SiLU (8 WMMA)
    __syncthreads();
    mlp_layer<HH, 3 * CC, 96, false>(bufA, sW3st, bufB); // 64 -> 96      (12 WMMA)
    __syncthreads();
    // Fused message scatter: A[rcv,lm,c] += sh[e,lm] * h1[snd,c] * w[e,lmap(lm),c] / 16
    if (tile < numTiles) {
      for (int rrow = 0; rrow < 16; ++rrow) {
        const int e = e0 + rrow;
        if (e >= EE) break;
        const int s = snd[e];
        const int rc = rcv[e];
        const float h1s = h1[s * CC + lane];
        float wsel[3];
        wsel[0] = (float)bufB[rrow * 96 + lane];
        wsel[1] = (float)bufB[rrow * 96 + 32 + lane];
        wsel[2] = (float)bufB[rrow * 96 + 64 + lane];
#pragma unroll
        for (int lm = 0; lm < NSH; ++lm) {
          const float val = shb[e * NSH + lm] * h1s * wsel[lmap_of(lm)] * (1.0f / 16.0f);
          atomicAdd(&Abuf[(rc * NSH + lm) * CC + lane], val);
        }
      }
    }
    __syncthreads();
  }
}

// ---------------- per-node update: product basis, poly, self-connection, readout ----------------
__global__ void __launch_bounds__(256)
k_node_update(const float* __restrict__ Abuf, const float* __restrict__ h_in,
              const float* __restrict__ na,
              const float* __restrict__ Wprod, const float* __restrict__ Wp,
              const float* __restrict__ Wsc,
              const float* __restrict__ wro1, const float* __restrict__ Wm1,
              const float* __restrict__ wm2,
              float* __restrict__ energy, float* __restrict__ feats,
              float* __restrict__ h_out, int iter) {
  const int lane = threadIdx.x & 31;
  const int wid = (blockIdx.x * blockDim.x + threadIdx.x) >> 5;
  const int nwaves = (gridDim.x * blockDim.x) >> 5;
  for (int n = wid; n < NN; n += nwaves) {
    int zn = 0; float best = -1.0f;
    for (int zq = 0; zq < ZZ; ++zq) {
      const float v = na[n * ZZ + zq];
      if (v > best) { best = v; zn = zq; }
    }
    float sc = 0.0f;
#pragma unroll
    for (int cc = 0; cc < CC; ++cc)
      sc += h_in[n * CC + cc] * Wsc[(zn * CC + cc) * CC + lane];
    float out0 = 0.0f;
    for (int m = 0; m < NSH; ++m) {
      const int lm = lmap_of(m);
      float o = 0.0f;
#pragma unroll
      for (int cc = 0; cc < CC; ++cc)
        o += Abuf[(n * NSH + m) * CC + cc] * Wprod[(lm * CC + cc) * CC + lane];
      if (m == 0) out0 = o;
      else feats[n * (2 * NSH * CC) + m * CC + lane] = o;
    }
    const float A0 = Abuf[(n * NSH) * CC + lane];
    float poly = 0.0f, p = A0;
#pragma unroll
    for (int k = 0; k < 3; ++k) {
      poly += Wp[(k * ZZ + zn) * CC + lane] * p;
      p *= A0;
    }
    const float hsc = out0 + poly + sc;
    feats[n * (2 * NSH * CC) + lane] = hsc;
    h_out[n * CC + lane] = hsc;
    float e;
    if (iter == 0) {
      e = warp_sum(hsc * wro1[lane]);
    } else {
      float esum = 0.0f;
      for (int j = 0; j < MLPHH; ++j) {
        const float t = warp_sum(hsc * Wm1[lane * MLPHH + j]);
        esum += fast_silu(t) * wm2[j];
      }
      e = esum;
    }
    if (lane == 0) energy[n] += e;
  }
}

extern "C" void kernel_launch(void* const* d_in, const int* in_sizes, int n_in,
                              void* d_out, int out_size, void* d_ws, size_t ws_size,
                              hipStream_t stream) {
  (void)in_sizes; (void)n_in; (void)out_size; (void)ws_size;
  const float* na    = (const float*)d_in[0];
  const float* pos   = (const float*)d_in[1];
  const int*   snd   = (const int*)d_in[2];
  const int*   rcv   = (const int*)d_in[3];
  const float* Wemb  = (const float*)d_in[4];
  const float* aE    = (const float*)d_in[5];
  const float* Wr0   = (const float*)d_in[6];
  const float* Wr1   = (const float*)d_in[7];
  const float* Wr2   = (const float*)d_in[8];
  const float* Wr3   = (const float*)d_in[9];
  const float* Wlin  = (const float*)d_in[10];
  const float* Wsc   = (const float*)d_in[11];
  const float* Wprod = (const float*)d_in[12];
  const float* Wp    = (const float*)d_in[13];
  const float* wro1  = (const float*)d_in[14];
  const float* Wm1   = (const float*)d_in[15];
  const float* wm2   = (const float*)d_in[16];

  float* out = (float*)d_out;
  float* ws  = (float*)d_ws;
  size_t off = 0;
  float* shb = ws + off; off += (size_t)EE * NSH;        // 7.2M
  float* Rb  = ws + off; off += (size_t)EE * NBF;        // 6.4M
  float* hE  = ws + off; off += (size_t)NN * CC;
  float* h1  = ws + off; off += (size_t)NN * CC;
  float* hS0 = ws + off; off += (size_t)NN * CC;
  float* hS1 = ws + off; off += (size_t)NN * CC;
  float* Ab  = ws + off; off += (size_t)NN * NSH * CC;   // 14.4M (57.6 MB, L2-resident)

  k_node_init<<<(NN + 255) / 256, 256, 0, stream>>>(na, Wemb, aE, out, hE);
  k_edge_geom<<<(EE + 255) / 256, 256, 0, stream>>>(na, pos, snd, rcv, shb, Rb, out);

  const float* hcur = hE;
  float* houts[2] = {hS0, hS1};
  const int numTiles = (EE + 15) / 16;
  const int numGroups = (numTiles + 3) / 4;
  for (int i = 0; i < 2; ++i) {
    hipMemsetAsync(Ab, 0, (size_t)NN * NSH * CC * sizeof(float), stream);
    k_h_lin<<<(NN * CC + 255) / 256, 256, 0, stream>>>(hcur, Wlin + i * CC * CC, h1);
    k_edge_mlp<<<numGroups, 128, 0, stream>>>(Rb, shb, snd, rcv, h1,
        Wr0 + i * NBF * HH, Wr1 + i * HH * HH, Wr2 + i * HH * HH,
        Wr3 + i * HH * 3 * CC, Ab);
    k_node_update<<<1024, 256, 0, stream>>>(Ab, hcur, na,
        Wprod + i * 3 * CC * CC, Wp + i * 3 * ZZ * CC, Wsc + i * ZZ * CC * CC,
        wro1, Wm1, wm2, out, out + NN + i * NSH * CC, houts[i], i);
    hcur = houts[i];
  }
}
